// KGEGraphEncoder_13434657701936
// MI455X (gfx1250) — matrix operand order repaired and checked by reading, not compiled, not run on hardware
//
#include <hip/hip_runtime.h>

typedef __bf16 bf16;
typedef __attribute__((ext_vector_type(8)))  bf16  v8bf;
typedef __attribute__((ext_vector_type(16))) bf16  v16bf;
typedef __attribute__((ext_vector_type(8)))  float v8f;

#define NNODES 50000
#define NEDGES 800000
#define NRELS  500
#define HID    128
#define NH     8
#define NLAYER 2
#define NB     64
#define SLOPE  0.2f

// ---- monotone float <-> uint encoding for atomicMax on floats ----
__device__ __forceinline__ unsigned encf(float f){
  unsigned u = __float_as_uint(f);
  return (u & 0x80000000u) ? ~u : (u | 0x80000000u);
}
__device__ __forceinline__ float decf(unsigned u){
  return (u & 0x80000000u) ? __uint_as_float(u & 0x7FFFFFFFu) : __uint_as_float(~u);
}

// ---------------- utility kernels ----------------
__global__ void k_zero_f32(float* p, int n){
  int i = blockIdx.x * 256 + threadIdx.x; if (i < n) p[i] = 0.0f;
}
__global__ void k_zero_u32(unsigned* p, int n){
  int i = blockIdx.x * 256 + threadIdx.x; if (i < n) p[i] = 0u;
}
__global__ void k_f32_to_bf16(const float* __restrict__ x, bf16* __restrict__ y, int n){
  int i = blockIdx.x * 256 + threadIdx.x; if (i < n) y[i] = (bf16)x[i];
}
// gather rows from a table and convert to bf16:  y[r,k] = tab[ids[r], k]
__global__ void k_gather_bf16(const float* __restrict__ tab, const int* __restrict__ ids,
                              bf16* __restrict__ y, int nrows){
  int i = blockIdx.x * 256 + threadIdx.x;
  if (i < nrows * HID){
    int r = i >> 7, k = i & 127;
    y[i] = (bf16)tab[(size_t)ids[r] * HID + k];
  }
}
// transpose-convert a [128(k),128(c)] f32 weight to [128(c),128(k)] bf16 (K-contiguous B operand)
__global__ void k_conv_w_t(const float* __restrict__ W, bf16* __restrict__ Wt){
  int i = blockIdx.x * 256 + threadIdx.x;           // 16384 threads
  int c = i >> 7, k = i & 127;
  Wt[c * HID + k] = (bf16)W[k * HID + c];
}

// ---------------- WMMA GEMM: C[M,128] = A[M,128] * B[128,128] ----------------
// A: bf16 row-major [Mrows,128].  Bt: bf16 [col,128] (K-contiguous per column).
// One wave -> one 16x16 tile (mtile, ntile); ntile == head index.
__global__ void k_gemm_bf16(const bf16* __restrict__ A, const bf16* __restrict__ Bt,
                            float* __restrict__ C, int Mrows){
  int wave = threadIdx.x >> 5;                    // 8 waves / block
  int lane = threadIdx.x & 31;
  int mtiles = (Mrows + 15) >> 4;
  int mtile = blockIdx.x * 8 + wave;
  if (mtile >= mtiles) return;                    // wave-uniform exit
  int ntile = blockIdx.y;                         // 0..7
  int hsel = lane >> 4;                           // lane half: 0 or 1
  int lm   = lane & 15;

  int rowA = mtile * 16 + lm; if (rowA > Mrows - 1) rowA = Mrows - 1;  // clamp (pad tiles)
  int colB = ntile * 16 + lm;
  const bf16* arow = A  + (size_t)rowA * HID;
  const bf16* brow = Bt + (size_t)colB * HID;

  v8f acc = {};
#pragma unroll
  for (int kc = 0; kc < 4; ++kc){
    // A fragment (16-bit 16x32 layout): half0 lanes: K = 0..7 & 16..23 ; half1: K = 8..15 & 24..31
    int kA = kc * 32 + (hsel ? 8 : 0);
    v8bf alo = *(const v8bf*)(arow + kA);
    v8bf ahi = *(const v8bf*)(arow + kA + 16);
    v16bf a;
#pragma unroll
    for (int i = 0; i < 8; ++i){ a[i] = alo[i]; a[8 + i] = ahi[i]; }
    // B fragment (32x16): half0 lanes: K = 0..7 & 16..23 ; half1: K = 8..15 & 24..31 ; N = lane%16
    int kB = kc * 32 + (hsel ? 8 : 0);
    v8bf blo = *(const v8bf*)(brow + kB);
    v8bf bhi = *(const v8bf*)(brow + kB + 16);
    v16bf b;
#pragma unroll
    for (int i = 0; i < 8; ++i){ b[i] = blo[i]; b[8 + i] = bhi[i]; }

    acc = __builtin_amdgcn_wmma_f32_16x16x32_bf16(false, a, false, b, (short)0, acc,
                                                  false, false);
  }
  // D layout: VGPR r -> row (r + 8*half), col = lane%16
  int rbase = mtile * 16 + (hsel ? 8 : 0);
  float* ccol = C + (size_t)ntile * 16 + lm;
  if (mtile * 16 + 16 <= Mrows) {
    // full tile: wave-uniform, unguarded stores (no exec-mask churn)
#pragma unroll
    for (int r = 0; r < 8; ++r)
      ccol[(size_t)(rbase + r) * HID] = acc[r];
  } else {
    // ragged last tile only (relation GEMM)
#pragma unroll
    for (int r = 0; r < 8; ++r){
      int row = rbase + r;
      if (row < Mrows) ccol[(size_t)row * HID] = acc[r];
    }
  }
}

// ---------------- attention score dots ----------------
__global__ void k_scores2(const float* __restrict__ feat, const float* __restrict__ al,
                          const float* __restrict__ ar, float* __restrict__ el,
                          float* __restrict__ er, int rows){
  int i = blockIdx.x * 256 + threadIdx.x;
  if (i >= rows * NH) return;
  int r = i >> 3, h = i & 7;
  const float* f = feat + (size_t)r * HID + h * 16;
  float sl = 0.f, sr = 0.f;
#pragma unroll
  for (int d = 0; d < 16; ++d){ sl += f[d] * al[h * 16 + d]; sr += f[d] * ar[h * 16 + d]; }
  el[i] = sl; er[i] = sr;
}
__global__ void k_scores1(const float* __restrict__ feat, const float* __restrict__ ae,
                          float* __restrict__ ee, int rows){
  int i = blockIdx.x * 256 + threadIdx.x;
  if (i >= rows * NH) return;
  int r = i >> 3, h = i & 7;
  const float* f = feat + (size_t)r * HID + h * 16;
  float s = 0.f;
#pragma unroll
  for (int d = 0; d < 16; ++d) s += f[d] * ae[h * 16 + d];
  ee[i] = s;
}

// ---------------- edge passes ----------------
__global__ void k_edge1(const int* __restrict__ src, const int* __restrict__ dst,
                        const int* __restrict__ rel, const float* __restrict__ el,
                        const float* __restrict__ er, const float* __restrict__ ee,
                        float* __restrict__ ebuf, unsigned* __restrict__ emax, int E){
  int e = blockIdx.x * 256 + threadIdx.x;
  if (e >= E) return;
  int s = src[e], d = dst[e], r = rel[e];
#pragma unroll
  for (int h = 0; h < NH; ++h){
    float v = el[s * NH + h] + er[d * NH + h] + ee[r * NH + h];
    v = v > 0.f ? v : SLOPE * v;                 // leaky relu
    ebuf[(size_t)e * NH + h] = v;
    atomicMax(&emax[d * NH + h], encf(v));
  }
}
__global__ void k_edge2(const int* __restrict__ dst, float* __restrict__ ebuf,
                        const unsigned* __restrict__ emax, float* __restrict__ denom, int E){
  int e = blockIdx.x * 256 + threadIdx.x;
  if (e >= E) return;
  int d = dst[e];
#pragma unroll
  for (int h = 0; h < NH; ++h){
    float m = decf(emax[d * NH + h]);
    float x = __expf(ebuf[(size_t)e * NH + h] - m);
    ebuf[(size_t)e * NH + h] = x;
    atomicAdd(&denom[d * NH + h], x);
  }
}
__global__ void k_edge3(const int* __restrict__ src, const int* __restrict__ dst,
                        const int* __restrict__ rel, const float* __restrict__ ebuf,
                        const float* __restrict__ denom, const float* __restrict__ feat,
                        const float* __restrict__ relproj, float* __restrict__ outF, int total){
  int i = blockIdx.x * 256 + threadIdx.x;
  if (i >= total) return;
  int e = i >> 3, h = i & 7;
  int s = src[e], d = dst[e], r = rel[e];
  float alpha = ebuf[i] / denom[d * NH + h];
  const float4* fs = (const float4*)(feat    + (size_t)s * HID + h * 16);
  const float4* fr = (const float4*)(relproj + (size_t)r * HID + h * 16);
  float* o = outF + (size_t)d * HID + h * 16;
#pragma unroll
  for (int q = 0; q < 4; ++q){
    float4 a = fs[q], b = fr[q];
    atomicAdd(o + q * 4 + 0, (a.x + b.x) * alpha);
    atomicAdd(o + q * 4 + 1, (a.y + b.y) * alpha);
    atomicAdd(o + q * 4 + 2, (a.z + b.z) * alpha);
    atomicAdd(o + q * 4 + 3, (a.w + b.w) * alpha);
  }
}

// ---------------- CLS readout ----------------
__global__ void k_cls(const float* __restrict__ hfin, const int* __restrict__ offs,
                      float* __restrict__ out, int n){
  int i = blockIdx.x * 256 + threadIdx.x;
  if (i < n){
    int b = i >> 7, k = i & 127;
    out[i] = hfin[(size_t)offs[b] * HID + k];
  }
}

extern "C" void kernel_launch(void* const* d_in, const int* in_sizes, int n_in,
                              void* d_out, int out_size, void* d_ws, size_t ws_size,
                              hipStream_t stream) {
  const float* ent_table = (const float*)d_in[0];
  const float* rel_table = (const float*)d_in[1];
  const float* W_ent     = (const float*)d_in[2];
  const float* W_rel     = (const float*)d_in[3];
  const float* attn_l    = (const float*)d_in[4];
  const float* attn_r    = (const float*)d_in[5];
  const float* attn_e    = (const float*)d_in[6];
  const int*   ent_ids   = (const int*)d_in[7];
  const int*   rel_ids   = (const int*)d_in[8];
  const int*   src       = (const int*)d_in[9];
  const int*   dst       = (const int*)d_in[10];
  const int*   goffs     = (const int*)d_in[11];
  float*       out       = (float*)d_out;

  // workspace carve-up (256B aligned slices)
  size_t off = 0;
  auto alloc = [&](size_t bytes) -> char* {
    char* p = (char*)d_ws + off;
    off += (bytes + 255) & ~(size_t)255;
    return p;
  };
  bf16*     hbf     = (bf16*)    alloc((size_t)NNODES * HID * 2);
  float*    feat    = (float*)   alloc((size_t)NNODES * HID * 4);
  float*    hout    = (float*)   alloc((size_t)NNODES * HID * 4);
  bf16*     relbf   = (bf16*)    alloc((size_t)NRELS  * HID * 2);
  bf16*     WtE     = (bf16*)    alloc((size_t)HID * HID * 2);
  bf16*     WtR     = (bf16*)    alloc((size_t)HID * HID * 2);
  float*    relproj = (float*)   alloc((size_t)NRELS  * HID * 4);
  float*    el      = (float*)   alloc((size_t)NNODES * NH * 4);
  float*    er      = (float*)   alloc((size_t)NNODES * NH * 4);
  float*    ee      = (float*)   alloc((size_t)NRELS  * NH * 4);
  float*    ebuf    = (float*)   alloc((size_t)NEDGES * NH * 4);
  unsigned* emax    = (unsigned*)alloc((size_t)NNODES * NH * 4);
  float*    denom   = (float*)   alloc((size_t)NNODES * NH * 4);

  auto nb = [](long n) -> unsigned { return (unsigned)((n + 255) / 256); };

  // one-time: bf16 copies of the (deduped) relation table and the gathered node features
  k_f32_to_bf16<<<nb(NRELS * HID), 256, 0, stream>>>(rel_table, relbf, NRELS * HID);
  k_gather_bf16<<<nb((long)NNODES * HID), 256, 0, stream>>>(ent_table, ent_ids, hbf, NNODES);

  for (int l = 0; l < NLAYER; ++l) {
    if (l > 0)
      k_f32_to_bf16<<<nb((long)NNODES * HID), 256, 0, stream>>>(hout, hbf, NNODES * HID);
    k_conv_w_t<<<nb(HID * HID), 256, 0, stream>>>(W_ent + (size_t)l * HID * HID, WtE);
    k_conv_w_t<<<nb(HID * HID), 256, 0, stream>>>(W_rel + (size_t)l * HID * HID, WtR);

    dim3 gN((3125 + 7) / 8, 8);                // 50000/16 = 3125 row tiles, 8 head tiles
    k_gemm_bf16<<<gN, 256, 0, stream>>>(hbf, WtE, feat, NNODES);
    dim3 gR((32 + 7) / 8, 8);                  // ceil(500/16)=32 row tiles
    k_gemm_bf16<<<gR, 256, 0, stream>>>(relbf, WtR, relproj, NRELS);

    k_scores2<<<nb((long)NNODES * NH), 256, 0, stream>>>(feat, attn_l + l * HID,
                                                         attn_r + l * HID, el, er, NNODES);
    k_scores1<<<nb(NRELS * NH), 256, 0, stream>>>(relproj, attn_e + l * HID, ee, NRELS);

    k_zero_u32<<<nb((long)NNODES * NH), 256, 0, stream>>>(emax, NNODES * NH);
    k_zero_f32<<<nb((long)NNODES * NH), 256, 0, stream>>>(denom, NNODES * NH);
    k_zero_f32<<<nb((long)NNODES * HID), 256, 0, stream>>>(hout, NNODES * HID);

    k_edge1<<<nb(NEDGES), 256, 0, stream>>>(src, dst, rel_ids, el, er, ee, ebuf, emax, NEDGES);
    k_edge2<<<nb(NEDGES), 256, 0, stream>>>(dst, ebuf, emax, denom, NEDGES);
    k_edge3<<<nb((long)NEDGES * NH), 256, 0, stream>>>(src, dst, rel_ids, ebuf, denom,
                                                       feat, relproj, hout, NEDGES * NH);
  }

  k_cls<<<nb(NB * HID), 256, 0, stream>>>(hout, goffs, out, NB * HID);
}
